// MultiHeadSelfAttention_56762287784156
// MI455X (gfx1250) — matrix operand order repaired and checked by reading, not compiled, not run on hardware
//
#include <hip/hip_runtime.h>
#include <cstdint>
#include <cstddef>

// ---------------------------------------------------------------------------
// MHSA for MI455X (gfx1250): bf16 WMMA everywhere, async global->LDS staging.
//   D_MODEL=1024, H=16, Dh=64, B=2, S=2048, SCALE = 1/sqrt(64) = 0.125
// ---------------------------------------------------------------------------

#define D_MODEL   1024
#define NUM_HEADS 16
#define HEAD_DIM  64
#define SEQ       2048
#define BATCH     2
#define ATT_SCALE 0.125f

typedef __attribute__((ext_vector_type(16))) __bf16 v16bf;
typedef __attribute__((ext_vector_type(8)))  __bf16 v8bf;
typedef __attribute__((ext_vector_type(8)))  float  v8f;
typedef __attribute__((ext_vector_type(4)))  int    v4i;

// ---- gfx1250 async global->LDS copies (ASYNCcnt-tracked) -------------------
#if defined(__has_builtin)
#  if __has_builtin(__builtin_amdgcn_global_load_async_to_lds_b128) && \
      __has_builtin(__builtin_amdgcn_s_wait_asynccnt)
#    define USE_ASYNC_COPY 1
#  endif
#endif
#ifndef USE_ASYNC_COPY
#  define USE_ASYNC_COPY 0
#endif

#if USE_ASYNC_COPY
typedef __attribute__((address_space(1))) v4i* gbl_v4i_ptr;
typedef __attribute__((address_space(3))) v4i* lds_v4i_ptr;
__device__ __forceinline__ void async_copy16(const void* gsrc, void* ldst) {
    __builtin_amdgcn_global_load_async_to_lds_b128(
        (gbl_v4i_ptr)(gsrc), (lds_v4i_ptr)(ldst), 0, 0);
}
__device__ __forceinline__ void async_copy_wait() {
    __builtin_amdgcn_s_wait_asynccnt(0);
}
#else
__device__ __forceinline__ void async_copy16(const void* gsrc, void* ldst) {
    *(uint4*)ldst = *(const uint4*)gsrc;
}
__device__ __forceinline__ void async_copy_wait() {}
#endif

__device__ __forceinline__ unsigned short f2bf(float f) {
    union { float f; unsigned u; } v; v.f = f;
    unsigned u = v.u;
    u += 0x7FFFu + ((u >> 16) & 1u);      // round-to-nearest-even
    return (unsigned short)(u >> 16);
}

__device__ __forceinline__ v16bf make_frag(v8bf lo, v8bf hi) {
    union { struct { v8bf lo, hi; } p; v16bf v; } u;
    u.p.lo = lo; u.p.hi = hi;
    return u.v;
}

// ---------------------------------------------------------------------------
// One-time conversions
// ---------------------------------------------------------------------------
__global__ __launch_bounds__(256)
void cvt_f32_bf16(const float* __restrict__ in, unsigned short* __restrict__ out) {
    int i = (blockIdx.x * 256 + threadIdx.x) * 4;
    float4 f = *(const float4*)(in + i);
    ushort4 o;
    o.x = f2bf(f.x); o.y = f2bf(f.y); o.z = f2bf(f.z); o.w = f2bf(f.w);
    *(ushort4*)(out + i) = o;
}

// W fp32 [K][N] -> WT bf16 [N][K]
__global__ __launch_bounds__(256)
void transpose_cvt(const float* __restrict__ W, unsigned short* __restrict__ WT,
                   int K, int N) {
    __shared__ float tile[32][33];
    const int k0 = blockIdx.x * 32;
    const int n0 = blockIdx.y * 32;
    const int tx = threadIdx.x & 31;
    const int ty = threadIdx.x >> 5;      // 0..7
    #pragma unroll
    for (int i = 0; i < 4; ++i)
        tile[ty + i * 8][tx] = W[(size_t)(k0 + ty + i * 8) * N + n0 + tx];
    __syncthreads();
    #pragma unroll
    for (int i = 0; i < 4; ++i)
        WT[(size_t)(n0 + ty + i * 8) * K + k0 + tx] = f2bf(tile[tx][ty + i * 8]);
}

// ---------------------------------------------------------------------------
// GEMM: C[M,N] = A[M,K] * WT[N,K]^T + bias[N]   (A, WT both bf16)
//   Block tile 128x64, K-step 64; 8 waves (4x2), 32x32 per wave,
//   8 x v_wmma_f32_16x16x32_bf16 per K-step. All staging via async b128.
//   OUT_MODE: 0 = bf16 [M][N], 1 = fp32 [M][N], 2 = bf16 VT [b][col][seq]
// ---------------------------------------------------------------------------
template<int OUT_MODE>
__global__ __launch_bounds__(256)
void gemm_bf16_wmma(const unsigned short* __restrict__ A,
                    const unsigned short* __restrict__ WT,
                    const float* __restrict__ bias,
                    void* __restrict__ outp,
                    int M, int N, int K)
{
    const int n0   = blockIdx.x * 64;
    const int m0   = blockIdx.y * 128;
    const int tid  = threadIdx.x;
    const int lane = tid & 31;
    const int wave = tid >> 5;        // 0..7
    const int wm   = wave >> 1;       // 0..3
    const int wn   = wave & 1;        // 0..1
    const int l16  = lane & 15;
    const int lhalf = lane >> 4;      // 0|1

    constexpr int LDA = 72;           // bf16 elems per LDS row (64 + pad)
    constexpr int LDB = 72;
    __shared__ __align__(16) unsigned short sA[128 * LDA];   // A tile 128x64
    __shared__ __align__(16) unsigned short sB[64  * LDB];   // WT tile [n][k]

    v8f acc[2][2] = {};

    for (int k0 = 0; k0 < K; k0 += 64) {
        // ---- stage A tile (128x64): 1024 16B chunks, 4 per thread ----
        #pragma unroll
        for (int i = 0; i < 4; ++i) {
            int g   = tid + i * 256;
            int row = g >> 3;
            int c8  = (g & 7) * 8;
            async_copy16(A + (size_t)(m0 + row) * K + k0 + c8,
                         &sA[row * LDA + c8]);
        }
        // ---- stage WT tile (64x64): 512 16B chunks, 2 per thread ----
        #pragma unroll
        for (int i = 0; i < 2; ++i) {
            int g  = tid + i * 256;
            int n  = g >> 3;
            int c8 = (g & 7) * 8;
            async_copy16(WT + (size_t)(n0 + n) * K + k0 + c8,
                         &sB[n * LDB + c8]);
        }
        async_copy_wait();
        __syncthreads();

        // ---- fragments + 8 WMMAs ----
        #pragma unroll
        for (int ks = 0; ks < 2; ++ks) {
            v16bf afrag[2], bfrag[2];
            #pragma unroll
            for (int fm = 0; fm < 2; ++fm) {
                int row  = wm * 32 + fm * 16 + l16;
                int koff = ks * 32 + lhalf * 8;     // A: lanes>=16 start at K=+8
                v8bf lo = *(const v8bf*)&sA[row * LDA + koff];
                v8bf hi = *(const v8bf*)&sA[row * LDA + koff + 16];
                afrag[fm] = make_frag(lo, hi);
            }
            #pragma unroll
            for (int fn = 0; fn < 2; ++fn) {
                int n    = wn * 32 + fn * 16 + l16;
                int koff = ks * 32 + lhalf * 8;     // B: lanes>=16 start at K=+8
                v8bf lo = *(const v8bf*)&sB[n * LDB + koff];
                v8bf hi = *(const v8bf*)&sB[n * LDB + koff + 16];
                bfrag[fn] = make_frag(lo, hi);
            }
            #pragma unroll
            for (int fm = 0; fm < 2; ++fm)
                #pragma unroll
                for (int fn = 0; fn < 2; ++fn)
                    acc[fm][fn] = __builtin_amdgcn_wmma_f32_16x16x32_bf16(
                        false, afrag[fm], false, bfrag[fn],
                        (short)0, acc[fm][fn], false, false);
        }
        __syncthreads();
    }

    // ---- epilogue: bias + store ----
    unsigned short* outB = (unsigned short*)outp;
    float*          outF = (float*)outp;
    #pragma unroll
    for (int fm = 0; fm < 2; ++fm)
        #pragma unroll
        for (int fn = 0; fn < 2; ++fn) {
            int col  = n0 + wn * 32 + fn * 16 + l16;
            float bv = bias ? bias[col] : 0.0f;
            #pragma unroll
            for (int r = 0; r < 8; ++r) {
                int row = m0 + wm * 32 + fm * 16 + r + lhalf * 8;
                float v = acc[fm][fn][r] + bv;
                if (OUT_MODE == 1) {
                    outF[(size_t)row * N + col] = v;
                } else if (OUT_MODE == 0) {
                    outB[(size_t)row * N + col] = f2bf(v);
                } else {                       // VT: [b][col][seq_row]
                    int bb = row >> 11;        // row / SEQ
                    int rs = row & (SEQ - 1);
                    outB[(size_t)bb * D_MODEL * SEQ + (size_t)col * SEQ + rs] = f2bf(v);
                }
            }
        }
}

// ---------------------------------------------------------------------------
// Flash attention: one block = 128 Q rows of one (b,h); one wave = 16 Q rows.
// K chunk [kv][d] and pre-transposed V chunk [d][kv] both staged with async
// b128 copies. 16 WMMAs per 64-kv chunk per wave.
// ---------------------------------------------------------------------------
__global__ __launch_bounds__(256)
void attn_wmma(const unsigned short* __restrict__ Q,
               const unsigned short* __restrict__ Kc,
               const unsigned short* __restrict__ VT,   // [b][d_model][seq]
               unsigned short* __restrict__ O)
{
    const int tid   = threadIdx.x;
    const int lane  = tid & 31;
    const int wave  = tid >> 5;       // 0..7
    const int l16   = lane & 15;
    const int lhalf = lane >> 4;

    const int bx = blockIdx.x;
    const int qt = bx & 15;           // S/128 = 16 tiles
    const int h  = (bx >> 4) & 15;
    const int b  = bx >> 8;

    constexpr int LDK  = 72;
    constexpr int LDVT = 72;
    constexpr int LDP  = 72;
    __shared__ __align__(16) unsigned short sK [64 * LDK];       // [kv][d]
    __shared__ __align__(16) unsigned short sVT[64 * LDVT];      // [d][kv]
    __shared__ __align__(16) unsigned short sP [8][16 * LDP];    // per wave

    const unsigned short* Qb = Q  + ((size_t)b * SEQ) * D_MODEL + (size_t)h * HEAD_DIM;
    const unsigned short* Kb = Kc + ((size_t)b * SEQ) * D_MODEL + (size_t)h * HEAD_DIM;
    const unsigned short* Vb = VT + (size_t)b * D_MODEL * SEQ + (size_t)(h * HEAD_DIM) * SEQ;

    const int qrow0 = qt * 128 + wave * 16;

    // Q fragments: registers for entire kernel (Dh=64 -> 2 frags)
    v16bf aq[2];
    {
        const unsigned short* qr = Qb + (size_t)(qrow0 + l16) * D_MODEL;
        #pragma unroll
        for (int ks = 0; ks < 2; ++ks) {
            int koff = ks * 32 + lhalf * 8;
            v8bf lo = *(const v8bf*)(qr + koff);
            v8bf hi = *(const v8bf*)(qr + koff + 16);
            aq[ks] = make_frag(lo, hi);
        }
    }

    float m_i[8], l_i[8];
    v8f o_acc[4] = {};
    #pragma unroll
    for (int r = 0; r < 8; ++r) { m_i[r] = -1e30f; l_i[r] = 0.0f; }

    for (int c = 0; c < SEQ / 64; ++c) {
        // ---- K chunk: 512 chunks, 2 per thread (async) ----
        #pragma unroll
        for (int p = 0; p < 2; ++p) {
            int slot = tid + p * 256;
            int kv   = slot >> 3;
            int d0   = (slot & 7) * 8;
            async_copy16(Kb + (size_t)(c * 64 + kv) * D_MODEL + d0,
                         &sK[kv * LDK + d0]);
        }
        // ---- V chunk (pre-transposed): 256 chunks, 1 per thread (async) ----
        {
            int d  = tid >> 2;            // 0..63
            int c8 = (tid & 3) * 16;      // 0..48, 8 bf16 per chunk? no: 16
            async_copy16(Vb + (size_t)d * SEQ + c * 64 + (tid & 3) * 16,
                         &sVT[d * LDVT + (tid & 3) * 16]);
            (void)c8;
        }
        async_copy_wait();
        __syncthreads();

        // ---- S = Q * K^T : four 16-col tiles, 2 K-steps over Dh ----
        v8f sacc[4] = {};
        #pragma unroll
        for (int fn = 0; fn < 4; ++fn) {
            #pragma unroll
            for (int ks = 0; ks < 2; ++ks) {
                int kvcol = fn * 16 + l16;
                int koff  = ks * 32 + lhalf * 8;
                v8bf lo = *(const v8bf*)&sK[kvcol * LDK + koff];
                v8bf hi = *(const v8bf*)&sK[kvcol * LDK + koff + 16];
                v16bf bfrag = make_frag(lo, hi);
                sacc[fn] = __builtin_amdgcn_wmma_f32_16x16x32_bf16(
                    false, aq[ks], false, bfrag, (short)0, sacc[fn], false, false);
            }
        }

        // ---- online softmax update (per lane: 8 rows, 4 cols) ----
        float pr[4][8];
        #pragma unroll
        for (int r = 0; r < 8; ++r) {
            float s0 = sacc[0][r] * ATT_SCALE;
            float s1 = sacc[1][r] * ATT_SCALE;
            float s2 = sacc[2][r] * ATT_SCALE;
            float s3 = sacc[3][r] * ATT_SCALE;
            float mx = fmaxf(fmaxf(s0, s1), fmaxf(s2, s3));
            #pragma unroll
            for (int msk = 1; msk < 16; msk <<= 1)
                mx = fmaxf(mx, __shfl_xor(mx, msk, 32));
            float mnew  = fmaxf(m_i[r], mx);
            float alpha = __expf(m_i[r] - mnew);
            float e0 = __expf(s0 - mnew);
            float e1 = __expf(s1 - mnew);
            float e2 = __expf(s2 - mnew);
            float e3 = __expf(s3 - mnew);
            float rs = (e0 + e1) + (e2 + e3);
            #pragma unroll
            for (int msk = 1; msk < 16; msk <<= 1)
                rs += __shfl_xor(rs, msk, 32);
            l_i[r] = l_i[r] * alpha + rs;
            m_i[r] = mnew;
            pr[0][r] = e0; pr[1][r] = e1; pr[2][r] = e2; pr[3][r] = e3;
            #pragma unroll
            for (int fo = 0; fo < 4; ++fo)
                o_acc[fo][r] *= alpha;
        }

        // ---- P: C-layout -> A-fragment layout via per-wave LDS tile ----
        {
            unsigned short* pw = &sP[wave][0];
            #pragma unroll
            for (int r = 0; r < 8; ++r) {
                int row = r + lhalf * 8;
                #pragma unroll
                for (int fn = 0; fn < 4; ++fn)
                    pw[row * LDP + fn * 16 + l16] = f2bf(pr[fn][r]);
            }
        }
        v16bf pfrag[2];
        {
            const unsigned short* pw = &sP[wave][0];
            #pragma unroll
            for (int kp = 0; kp < 2; ++kp) {
                int koff = kp * 32 + lhalf * 8;
                v8bf lo = *(const v8bf*)(pw + l16 * LDP + koff);
                v8bf hi = *(const v8bf*)(pw + l16 * LDP + koff + 16);
                pfrag[kp] = make_frag(lo, hi);
            }
        }

        // ---- O += P * V : 4 N-fragments over Dh, 2 K-steps over kv ----
        #pragma unroll
        for (int fo = 0; fo < 4; ++fo) {
            int d = fo * 16 + l16;
            #pragma unroll
            for (int kp = 0; kp < 2; ++kp) {
                int koff = kp * 32 + lhalf * 8;
                v8bf lo = *(const v8bf*)&sVT[d * LDVT + koff];
                v8bf hi = *(const v8bf*)&sVT[d * LDVT + koff + 16];
                v16bf vfrag = make_frag(lo, hi);
                o_acc[fo] = __builtin_amdgcn_wmma_f32_16x16x32_bf16(
                    false, pfrag[kp], false, vfrag, (short)0, o_acc[fo], false, false);
            }
        }
        __syncthreads();
    }

    // ---- normalize and store bf16 to attention-output workspace ----
    unsigned short* Ob = O + ((size_t)b * SEQ) * D_MODEL + (size_t)h * HEAD_DIM;
    #pragma unroll
    for (int fo = 0; fo < 4; ++fo) {
        int d = fo * 16 + l16;
        #pragma unroll
        for (int r = 0; r < 8; ++r) {
            int row = qrow0 + r + lhalf * 8;
            float v = o_acc[fo][r] / l_i[r];
            Ob[(size_t)row * D_MODEL + d] = f2bf(v);
        }
    }
}

// ---------------------------------------------------------------------------
// Launch
// ---------------------------------------------------------------------------
extern "C" void kernel_launch(void* const* d_in, const int* in_sizes, int n_in,
                              void* d_out, int out_size, void* d_ws, size_t ws_size,
                              hipStream_t stream)
{
    const float* query = (const float*)d_in[0];
    const float* key   = (const float*)d_in[1];
    const float* value = (const float*)d_in[2];
    const float* Wq    = (const float*)d_in[3];
    const float* bq    = (const float*)d_in[4];
    const float* Wk    = (const float*)d_in[5];
    const float* bk    = (const float*)d_in[6];
    const float* Wv    = (const float*)d_in[7];
    const float* bv    = (const float*)d_in[8];
    const float* Wo    = (const float*)d_in[9];
    const float* bo    = (const float*)d_in[10];
    float* out = (float*)d_out;

    const int M = BATCH * SEQ;            // 4096
    const int N = D_MODEL;                // 1024
    const int K = D_MODEL;                // 1024

    const size_t mn = (size_t)M * N;      // 4M elements
    const size_t kk = (size_t)K * N;      // 1M elements
    unsigned short* p = (unsigned short*)d_ws;
    unsigned short* Qi  = p; p += mn;     // bf16 inputs
    unsigned short* Ki  = p; p += mn;
    unsigned short* Vi  = p; p += mn;
    unsigned short* WqT = p; p += kk;     // bf16 transposed weights [N][K]
    unsigned short* WkT = p; p += kk;
    unsigned short* WvT = p; p += kk;
    unsigned short* WoT = p; p += kk;
    unsigned short* Qp  = p; p += mn;     // projections
    unsigned short* Kp  = p; p += mn;
    unsigned short* Vt  = p; p += mn;     // V projection, [b][col][seq]
    unsigned short* Aw  = p; p += mn;     // attention output

    // one-time conversions
    cvt_f32_bf16<<<(int)(mn / 1024), 256, 0, stream>>>(query, Qi);
    cvt_f32_bf16<<<(int)(mn / 1024), 256, 0, stream>>>(key,   Ki);
    cvt_f32_bf16<<<(int)(mn / 1024), 256, 0, stream>>>(value, Vi);
    dim3 tg(K / 32, N / 32);
    transpose_cvt<<<tg, 256, 0, stream>>>(Wq, WqT, K, N);
    transpose_cvt<<<tg, 256, 0, stream>>>(Wk, WkT, K, N);
    transpose_cvt<<<tg, 256, 0, stream>>>(Wv, WvT, K, N);
    transpose_cvt<<<tg, 256, 0, stream>>>(Wo, WoT, K, N);

    dim3 gblk(N / 64, M / 128);           // 16 x 32 blocks
    gemm_bf16_wmma<0><<<gblk, 256, 0, stream>>>(Qi, WqT, bq, Qp, M, N, K);
    gemm_bf16_wmma<0><<<gblk, 256, 0, stream>>>(Ki, WkT, bk, Kp, M, N, K);
    gemm_bf16_wmma<2><<<gblk, 256, 0, stream>>>(Vi, WvT, bv, Vt, M, N, K);

    attn_wmma<<<BATCH * NUM_HEADS * (SEQ / 128), 256, 0, stream>>>(Qp, Kp, Vt, Aw);

    gemm_bf16_wmma<1><<<gblk, 256, 0, stream>>>(Aw, WoT, bo, out, M, N, K);
}
